// DistillationTrainer_62448824484606
// MI455X (gfx1250) — matrix operand order repaired and checked by reading, not compiled, not run on hardware
//
#include <hip/hip_runtime.h>

#define H_  8
#define Q_  1024
#define N_  8192
#define NC_ 1024
#define D_  128

typedef __attribute__((ext_vector_type(16))) __bf16 v16bf;
typedef __attribute__((ext_vector_type(8)))  __bf16 v8bf;
typedef __attribute__((ext_vector_type(8)))  float  v8f;

// raw IEEE maximumNumber without clang's canonicalize ops (inputs are WMMA/exp
// results, already canonical)
__device__ __forceinline__ float fmax_raw(float a, float b) {
  float r;
  asm("v_max_num_f32 %0, %1, %2" : "=v"(r) : "v"(a), "v"(b));
  return r;
}

// --- row-max across the 16-lane halves of a wave32 via ds_swizzle xor patterns ---
__device__ __forceinline__ float grp16_max(float v) {
  v = fmax_raw(v, __int_as_float(__builtin_amdgcn_ds_swizzle(__float_as_int(v), 0x041f))); // xor 1
  v = fmax_raw(v, __int_as_float(__builtin_amdgcn_ds_swizzle(__float_as_int(v), 0x081f))); // xor 2
  v = fmax_raw(v, __int_as_float(__builtin_amdgcn_ds_swizzle(__float_as_int(v), 0x101f))); // xor 4
  v = fmax_raw(v, __int_as_float(__builtin_amdgcn_ds_swizzle(__float_as_int(v), 0x201f))); // xor 8
  return v;
}

// ---- fp32 -> bf16 row-major convert (8 elems/thread) ----
__global__ void cvt_bf16_kernel(const float* __restrict__ in, __bf16* __restrict__ out) {
  size_t i = ((size_t)blockIdx.x * blockDim.x + threadIdx.x) * 8;
  const float4* p = (const float4*)(in + i);
  float4 a = p[0], b = p[1];
  v8bf o;
  o[0] = (__bf16)a.x; o[1] = (__bf16)a.y; o[2] = (__bf16)a.z; o[3] = (__bf16)a.w;
  o[4] = (__bf16)b.x; o[5] = (__bf16)b.y; o[6] = (__bf16)b.z; o[7] = (__bf16)b.w;
  *(v8bf*)(out + i) = o;
}

// ---- [h][n][d] fp32 -> [h][d][n] bf16 (LDS-tiled transpose) ----
__global__ void transpose_bf16_kernel(const float* __restrict__ in, __bf16* __restrict__ out, int n) {
  __shared__ float tile[32][33];
  int h = blockIdx.z;
  int n0 = blockIdx.x * 32, d0 = blockIdx.y * 32;
  const float* src = in + (size_t)h * n * D_;
  __bf16* dst = out + (size_t)h * D_ * n;
  for (int r = threadIdx.y; r < 32; r += 8)
    tile[r][threadIdx.x] = src[(size_t)(n0 + r) * D_ + d0 + threadIdx.x];
  __syncthreads();
  for (int r = threadIdx.y; r < 32; r += 8)
    dst[(size_t)(d0 + r) * n + n0 + threadIdx.x] = (__bf16)tile[threadIdx.x][r];
}

// ---- async stage one 32-key tile (K: 32x128 bf16, Vt: 128x32 bf16) into LDS ----
__device__ __forceinline__ void issue_async_tile(const __bf16* __restrict__ Kg,
                                                 const __bf16* __restrict__ Vtg,
                                                 int nKeys, int kb,
                                                 __bf16* kbuf, __bf16* vbuf, int tid) {
#pragma unroll
  for (int j = 0; j < 2; ++j) {               // K tile: 32 rows x 256B
    int idx = tid * 2 + j;                    // 0..511 chunks of 16B
    int row = idx >> 4, chunk = idx & 15;
    const __bf16* g = Kg + (size_t)(kb + row) * D_ + chunk * 8;
    unsigned int l = (unsigned int)(size_t)(kbuf + row * D_ + chunk * 8);
    asm volatile("global_load_async_to_lds_b128 %0, %1, off"
                 :: "v"(l), "v"((unsigned long long)(size_t)g) : "memory");
  }
#pragma unroll
  for (int j = 0; j < 2; ++j) {               // Vt tile: 128 rows x 64B
    int idx = tid * 2 + j;
    int row = idx >> 2, chunk = idx & 3;
    const __bf16* g = Vtg + (size_t)row * nKeys + kb + chunk * 8;
    unsigned int l = (unsigned int)(size_t)(vbuf + row * 32 + chunk * 8);
    asm volatile("global_load_async_to_lds_b128 %0, %1, off"
                 :: "v"(l), "v"((unsigned long long)(size_t)g) : "memory");
  }
}

// ---- one flash-attention pass: 16 query rows/wave, 32-key tiles ----
// Softmax runs in the exp2 domain (log2(e) folded into the Q scale).
// O[0..7]: 16x128 output accumulator; O[8]: denominator (ones-column WMMA).
__device__ void flash_pass(const v16bf* aQ, v16bf bOnes, const __bf16* __restrict__ Kg,
                           const __bf16* __restrict__ Vtg, int nKeys,
                           __bf16* Kbuf0, __bf16* Kbuf1, __bf16* Vbuf0, __bf16* Vbuf1,
                           __bf16* ldsP, int tid, int lane, v8f* O) {
  const int laneLo = lane & 15;
  const int laneHi = lane >> 4;
  float mrow[8];
#pragma unroll
  for (int r = 0; r < 8; ++r) mrow[r] = -3.0e30f;
#pragma unroll
  for (int t = 0; t < 9; ++t)
#pragma unroll
    for (int r = 0; r < 8; ++r) O[t][r] = 0.f;

  // prologue: stage tile 0
  issue_async_tile(Kg, Vtg, nKeys, 0, Kbuf0, Vbuf0, tid);
  asm volatile("s_wait_asynccnt 0" ::: "memory");
  __syncthreads();

  for (int kb = 0; kb < nKeys; kb += 32) {
    const int cur = (kb >> 5) & 1;
    __bf16* kcur = cur ? Kbuf1 : Kbuf0;
    __bf16* vcur = cur ? Vbuf1 : Vbuf0;
    if (kb + 32 < nKeys)   // prefetch next tile while computing this one
      issue_async_tile(Kg, Vtg, nKeys, kb + 32, cur ? Kbuf0 : Kbuf1,
                       cur ? Vbuf0 : Vbuf1, tid);

    // S(16x32) = Q(16x128) x K^T, 2-deep register pipeline on the B-fragments
    v8f S0 = {0.f,0.f,0.f,0.f,0.f,0.f,0.f,0.f};
    v8f S1 = {0.f,0.f,0.f,0.f,0.f,0.f,0.f,0.f};
    v16bf kA0 = *(const v16bf*)(kcur + (size_t)laneLo * D_ + laneHi * 16);
    v16bf kA1 = *(const v16bf*)(kcur + (size_t)(16 + laneLo) * D_ + laneHi * 16);
#pragma unroll
    for (int c = 0; c < 4; ++c) {
      v16bf kB0, kB1;
      if (c < 3) {
        kB0 = *(const v16bf*)(kcur + (size_t)laneLo * D_ + (c + 1) * 32 + laneHi * 16);
        kB1 = *(const v16bf*)(kcur + (size_t)(16 + laneLo) * D_ + (c + 1) * 32 + laneHi * 16);
      }
      S0 = __builtin_amdgcn_wmma_f32_16x16x32_bf16(false, aQ[c], false, kA0, (short)0, S0, false, false);
      S1 = __builtin_amdgcn_wmma_f32_16x16x32_bf16(false, aQ[c], false, kA1, (short)0, S1, false, false);
      if (c < 3) { kA0 = kB0; kA1 = kB1; }
    }

    // online softmax (exp2 domain): row max via ds_swizzle, rescale only when
    // the running max actually changes (wave-uniform vote; rare after warmup)
    float rm[8];
#pragma unroll
    for (int r = 0; r < 8; ++r) rm[r] = grp16_max(fmax_raw(S0[r], S1[r]));
    int upd = 0;
#pragma unroll
    for (int r = 0; r < 8; ++r) upd |= (rm[r] > mrow[r]) ? 1 : 0;
    if (__any(upd)) {
#pragma unroll
      for (int r = 0; r < 8; ++r) {
        float mn = fmax_raw(mrow[r], rm[r]);
        float sc = exp2f(mrow[r] - mn);
        mrow[r] = mn;
#pragma unroll
        for (int t = 0; t < 9; ++t) O[t][r] *= sc;
      }
    }
    float p0[8], p1[8];
#pragma unroll
    for (int r = 0; r < 8; ++r) {
      p0[r] = exp2f(S0[r] - mrow[r]);
      p1[r] = exp2f(S1[r] - mrow[r]);
    }

    // C-layout -> LDS (row-major 16x32 bf16) -> A-layout fragment for PV
#pragma unroll
    for (int r = 0; r < 8; ++r) {
      int row = laneHi * 8 + r;
      ldsP[row * 32 + laneLo]      = (__bf16)p0[r];
      ldsP[row * 32 + 16 + laneLo] = (__bf16)p1[r];
    }
    asm volatile("s_wait_dscnt 0" ::: "memory");  // cross-lane LDS RAW inside the wave
    v16bf aP;
    {
      v8bf lo = *(const v8bf*)(ldsP + laneLo * 32 + laneHi * 16);
      v8bf hi = *(const v8bf*)(ldsP + laneLo * 32 + laneHi * 16 + 8);
#pragma unroll
      for (int i = 0; i < 8; ++i) { aP[i] = lo[i]; aP[8 + i] = hi[i]; }
    }

    // O(16x128) += P(16x32) x V(32x128); 2-deep pipeline; +1 WMMA for the row sums
    v16bf vA = *(const v16bf*)(vcur + (size_t)laneLo * 32 + laneHi * 16);
#pragma unroll
    for (int t = 0; t < 8; ++t) {
      v16bf vB;
      if (t < 7) vB = *(const v16bf*)(vcur + (size_t)((t + 1) * 16 + laneLo) * 32 + laneHi * 16);
      O[t] = __builtin_amdgcn_wmma_f32_16x16x32_bf16(false, aP, false, vA, (short)0, O[t], false, false);
      if (t < 7) vA = vB;
    }
    O[8] = __builtin_amdgcn_wmma_f32_16x16x32_bf16(false, aP, false, bOnes, (short)0, O[8], false, false);

    // finish the prefetch of tile t+1, then release buffers block-wide
    asm volatile("s_wait_asynccnt 0" ::: "memory");
    __syncthreads();
  }
#pragma unroll
  for (int r = 0; r < 8; ++r) {
    float inv = 1.0f / O[8][r];     // denominator from the ones-column WMMA
#pragma unroll
    for (int t = 0; t < 8; ++t) O[t][r] *= inv;
  }
}

__global__ __launch_bounds__(256, 1) void flash_mse_kernel(
    const float* __restrict__ Qf, const __bf16* __restrict__ Kb,
    const __bf16* __restrict__ Vt, const __bf16* __restrict__ cKb,
    const __bf16* __restrict__ cVt, float* __restrict__ partial) {
  __shared__ alignas(16) __bf16 Kbuf[2][32 * 128];   // 2 x 8KB double buffer
  __shared__ alignas(16) __bf16 Vbuf[2][128 * 32];   // 2 x 8KB (transposed tile)
  __shared__ alignas(16) __bf16 ldsP[8 * 16 * 32];   // 1KB/wave P-staging
  __shared__ float red[8];
  const int tid  = threadIdx.x;
  const int wave = tid >> 5;
  const int lane = tid & 31;
  const int laneLo = lane & 15, laneHi = lane >> 4;
  const int h  = blockIdx.x >> 3;
  const int qt = blockIdx.x & 7;
  const int row0 = qt * 128 + wave * 16;

  // Q A-fragments pre-scaled by log2(e)/sqrt(D): softmax runs in exp2 domain.
  const float scale = 0.08838834764831845f * 1.4426950408889634f;
  const float* Qrow = Qf + ((size_t)h * Q_ + row0 + laneLo) * D_;
  v16bf aQ[4];
#pragma unroll
  for (int c = 0; c < 4; ++c) {
#pragma unroll
    for (int i = 0; i < 16; ++i) {
      int k = c * 32 + laneHi * 16 + i;
      aQ[c][i] = (__bf16)(Qrow[k] * scale);
    }
  }
  v16bf bOnes;
#pragma unroll
  for (int i = 0; i < 16; ++i) bOnes[i] = (__bf16)1.0f;

  v8f Ot[9], Oc[9];
  flash_pass(aQ, bOnes, Kb  + (size_t)h * N_  * D_, Vt  + (size_t)h * D_ * N_,  N_,
             Kbuf[0], Kbuf[1], Vbuf[0], Vbuf[1], ldsP + wave * 512, tid, lane, Ot);
  flash_pass(aQ, bOnes, cKb + (size_t)h * NC_ * D_, cVt + (size_t)h * D_ * NC_, NC_,
             Kbuf[0], Kbuf[1], Vbuf[0], Vbuf[1], ldsP + wave * 512, tid, lane, Oc);

  float acc = 0.f;
#pragma unroll
  for (int t = 0; t < 8; ++t)
#pragma unroll
    for (int r = 0; r < 8; ++r) { float d = Ot[t][r] - Oc[t][r]; acc += d * d; }
#pragma unroll
  for (int m = 1; m < 32; m <<= 1) acc += __shfl_xor(acc, m, 32);
  if (lane == 0) red[wave] = acc;
  __syncthreads();
  if (tid == 0) {
    float s = 0.f;
#pragma unroll
    for (int w = 0; w < 8; ++w) s += red[w];
    partial[blockIdx.x] = s;
  }
}

__global__ void final_reduce_kernel(const float* __restrict__ partial, float* __restrict__ out) {
  if (threadIdx.x == 0) {
    float s = 0.f;
    for (int i = 0; i < 64; ++i) s += partial[i];   // fixed order -> deterministic
    out[0] = s * (1.0f / ((float)H_ * Q_ * D_));
  }
}

extern "C" void kernel_launch(void* const* d_in, const int* in_sizes, int n_in,
                              void* d_out, int out_size, void* d_ws, size_t ws_size,
                              hipStream_t stream) {
  (void)in_sizes; (void)n_in; (void)out_size; (void)ws_size;
  const float* q  = (const float*)d_in[0];
  const float* k  = (const float*)d_in[1];
  const float* v  = (const float*)d_in[2];
  const float* ck = (const float*)d_in[3];
  const float* cv = (const float*)d_in[4];

  char* ws = (char*)d_ws;
  const size_t szK  = (size_t)H_ * N_  * D_ * 2;  // 16 MiB
  const size_t szCK = (size_t)H_ * NC_ * D_ * 2;  //  2 MiB
  __bf16* Kb   = (__bf16*)(ws);
  __bf16* Vt   = (__bf16*)(ws + szK);
  __bf16* cKb  = (__bf16*)(ws + 2 * szK);
  __bf16* cVt  = (__bf16*)(ws + 2 * szK + szCK);
  float*  part = (float*) (ws + 2 * szK + 2 * szCK);

  cvt_bf16_kernel<<<(int)(((size_t)H_ * N_  * D_) / 2048), 256, 0, stream>>>(k,  Kb);
  cvt_bf16_kernel<<<(int)(((size_t)H_ * NC_ * D_) / 2048), 256, 0, stream>>>(ck, cKb);
  transpose_bf16_kernel<<<dim3(N_  / 32, D_ / 32, H_), dim3(32, 8), 0, stream>>>(v,  Vt,  N_);
  transpose_bf16_kernel<<<dim3(NC_ / 32, D_ / 32, H_), dim3(32, 8), 0, stream>>>(cv, cVt, NC_);
  flash_mse_kernel<<<H_ * (Q_ / 128), 256, 0, stream>>>(q, Kb, Vt, cKb, cVt, part);
  final_reduce_kernel<<<1, 32, 0, stream>>>(part, (float*)d_out);
}
